// SimpleSelfAttention_36627481101227
// MI455X (gfx1250) — matrix operand order repaired and checked by reading, not compiled, not run on hardware
//
#include <hip/hip_runtime.h>

// ---------------------------------------------------------------------------
// Self-attention for MI455X (gfx1250, wave32, WMMA).
// Pipeline: [1] QKV projection (f32 in -> bf16 out, V transposed)
//           [2] scores = Q K^T * 1/sqrt(D) + mask  (bf16 WMMA, f32 out)
//           [3] row softmax in place (f32)
//           [4] out = P V  (P converted to bf16 on the fly, bf16 WMMA)
// All GEMMs use v_wmma_f32_16x16x32_bf16, software-pipelined with the K-loop
// unrolled x2 so every ping/pong buffer index is a compile-time constant
// (prevents local-array scratch spills):
//   - bf16 tiles: double-buffered GLOBAL_LOAD_ASYNC_TO_LDS_B128 (ASYNCcnt)
//   - f32 tiles : register-pipelined loads + v_cvt_pk_bf16_f32 staging
// 192MB L2 keeps the 64MB score matrix + 48MB bf16 QKV workspace resident,
// so no flash-attention machinery is needed. Workspace requirement: 112 MB.
// ---------------------------------------------------------------------------

#define BATCH 4
#define SEQ   2048
#define DIM   1024

typedef __attribute__((ext_vector_type(16))) __bf16 v16bf;
typedef __attribute__((ext_vector_type(8)))  __bf16 v8bf;
typedef __attribute__((ext_vector_type(8)))  float  v8f;

// --- CDNA5 async copy: global -> LDS, 16 bytes per lane -------------------
// LDS dest address = low 32 bits of the generic pointer (ISA 10.2).
// Tracked by ASYNCcnt.
static __device__ inline void async_g2l_b128(void* lds_dst, const void* gsrc) {
  unsigned lds = (unsigned)(uintptr_t)lds_dst;
  unsigned long long ga = (unsigned long long)(uintptr_t)gsrc;
  asm volatile("global_load_async_to_lds_b128 %0, %1, off"
               :: "v"(lds), "v"(ga) : "memory");
}
static __device__ inline void wait_async0() {
  asm volatile("s_wait_asynccnt 0x0" ::: "memory");
}
static __device__ inline void wait_async2() {   // allow 2 newest outstanding
  asm volatile("s_wait_asynccnt 0x2" ::: "memory");
}
static __device__ inline void wait_async4() {   // allow 4 newest outstanding
  asm volatile("s_wait_asynccnt 0x4" ::: "memory");
}

// A fragment: 16x32 bf16 (M x K).  CDNA5 ISA §7.12.2:
//  lanes 0..15 : row M=lane, K = {0..7} in v0..3, {16..23} in v4..7
//  lanes 16..31: row M=lane-16, K = {8..15} in v0..3, {24..31} in v4..7
static __device__ inline v16bf load_a_frag(const __bf16* tile, int stride) {
  const int lane = threadIdx.x & 31;
  const int r  = lane & 15;
  const int kh = lane >> 4;
  const __bf16* p = tile + r * stride + kh * 8;
  v8bf lo = *(const v8bf*)(p);        // K = kh*8 .. kh*8+7
  v8bf hi = *(const v8bf*)(p + 16);   // K = 16+kh*8 .. 16+kh*8+7
  v16bf a;
#pragma unroll
  for (int i = 0; i < 8; ++i) { a[i] = lo[i]; a[i + 8] = hi[i]; }
  return a;
}

// B fragment: 32x16 bf16 (K x N). lanes 0..15: col N=lane, K={0..7,16..23};
// lanes 16..31: col N=lane-16, K={8..15,24..31}.  tile layout: [n][k], K contiguous.
static __device__ inline v16bf load_b_frag(const __bf16* tile, int stride) {
  const int lane = threadIdx.x & 31;
  const int n  = lane & 15;
  const int kg = lane >> 4;
  const __bf16* p = tile + n * stride + kg * 8;
  v8bf lo = *(const v8bf*)(p);
  v8bf hi = *(const v8bf*)(p + 16);
  v16bf b;
#pragma unroll
  for (int i = 0; i < 8; ++i) { b[i] = lo[i]; b[i + 8] = hi[i]; }
  return b;
}

static __device__ inline v8f zero8() {
  v8f z;
#pragma unroll
  for (int i = 0; i < 8; ++i) z[i] = 0.0f;
  return z;
}

// One K-step: A frag for this wave's 16 rows, 4 B frags (64 cols), 4 WMMAs.
static __device__ inline void mma_step(v8f* acc, const __bf16* Asub,
                                       const __bf16* Bbase) {
  v16bf a = load_a_frag(Asub, 32);
#pragma unroll
  for (int t = 0; t < 4; ++t) {
    v16bf b = load_b_frag(Bbase + (t * 16) * 32, 32);
    acc[t] = __builtin_amdgcn_wmma_f32_16x16x32_bf16(
        false, a, false, b, (short)0, acc[t], false, false);
  }
}

// Convert 4x float4 (16 f32) -> 16 bf16 into LDS.
static __device__ inline void cvt_store16(__bf16* dst, const float4* f) {
#pragma unroll
  for (int i = 0; i < 4; ++i) {
    dst[i * 4 + 0] = (__bf16)f[i].x;
    dst[i * 4 + 1] = (__bf16)f[i].y;
    dst[i * 4 + 2] = (__bf16)f[i].z;
    dst[i * 4 + 3] = (__bf16)f[i].w;
  }
}

static __device__ inline void load16(float4* dst, const float* src) {
#pragma unroll
  for (int i = 0; i < 4; ++i) dst[i] = *(const float4*)(src + i * 4);
}

// ---------------------------------------------------------------------------
// Kernel 1: Q/K/V = x @ W^T + b   (blockIdx.z selects which matrix)
// Q,K stored bf16 row-major [flat_row][e]; V stored transposed [b][e][s].
// f32 sources register-pipelined (static ping/pong): loads for tile i+1
// overlap WMMA of tile i.
// ---------------------------------------------------------------------------
__global__ __launch_bounds__(128) void qkv_kernel(
    const float* __restrict__ x,
    const float* __restrict__ wq, const float* __restrict__ bq,
    const float* __restrict__ wk, const float* __restrict__ bk,
    const float* __restrict__ wv, const float* __restrict__ bv,
    __bf16* __restrict__ Qb, __bf16* __restrict__ Kb, __bf16* __restrict__ Vt)
{
  __shared__ __attribute__((aligned(32))) __bf16 As[64 * 32];
  __shared__ __attribute__((aligned(32))) __bf16 Bs[64 * 32];

  const int which = blockIdx.z;
  const float* w    = (which == 0) ? wq : (which == 1) ? wk : wv;
  const float* bias = (which == 0) ? bq : (which == 1) ? bk : bv;

  const int rowBase = blockIdx.x * 64;   // flat row in [0, BATCH*SEQ)
  const int colBase = blockIdx.y * 64;   // output feature e
  const int tid  = threadIdx.x;
  const int wave = tid >> 5;
  const int lane = tid & 31;

  v8f acc[4];
#pragma unroll
  for (int t = 0; t < 4; ++t) acc[t] = zero8();

  const int r  = tid >> 1;          // 0..63 (staging row)
  const int c0 = (tid & 1) * 16;    // 0 or 16 (staging half-row)
  const float* baseA = x + (size_t)(rowBase + r) * DIM + c0;
  const float* baseB = w + (size_t)(colBase + r) * DIM + c0;
  __bf16* dA = As + r * 32 + c0;
  __bf16* dB = Bs + r * 32 + c0;
  const __bf16* Asub = As + (wave * 16) * 32;

  float4 pA[4], pB[4];               // ping registers (even tiles)
  float4 qA[4], qB[4];               // pong registers (odd tiles)
  load16(pA, baseA);                 // prologue: tile 0
  load16(pB, baseB);

  const int NT = DIM / 32;           // 32 K-steps (even)
  for (int it = 0; it < NT; it += 2) {
    // ---- even tile: consume ping, prefetch tile it+1 into pong ----
    load16(qA, baseA + (it + 1) * 32);         // it+1 < NT always (NT even)
    load16(qB, baseB + (it + 1) * 32);
    cvt_store16(dA, pA);
    cvt_store16(dB, pB);
    __syncthreads();
    mma_step(acc, Asub, Bs);
    __syncthreads();

    // ---- odd tile: consume pong, prefetch tile it+2 into ping ----
    if (it + 2 < NT) {
      load16(pA, baseA + (it + 2) * 32);
      load16(pB, baseB + (it + 2) * 32);
    }
    cvt_store16(dA, qA);
    cvt_store16(dB, qB);
    __syncthreads();
    mma_step(acc, Asub, Bs);
    __syncthreads();
  }

  // Epilogue: bias add + bf16 store. C/D layout: N = lane&15, M = i+8*(lane>>4).
  const int n  = lane & 15;
  const int kh = lane >> 4;
  const int row0 = rowBase + wave * 16;

  if (which == 2) {                        // V: store transposed [b][e][s]
#pragma unroll
    for (int t = 0; t < 4; ++t) {
      const int e  = colBase + t * 16 + n;
      const float be = bias[e];
      __bf16* col = Vt + ((size_t)(row0 >> 11)) * DIM * SEQ + (size_t)e * SEQ;
#pragma unroll
      for (int i = 0; i < 8; ++i) {
        const int s_ = (row0 & (SEQ - 1)) + i + 8 * kh;
        col[s_] = (__bf16)(acc[t][i] + be);
      }
    }
  } else {
    __bf16* dst = (which == 0) ? Qb : Kb;  // row-major [flat_row][e]
#pragma unroll
    for (int t = 0; t < 4; ++t) {
      const int e  = colBase + t * 16 + n;
      const float be = bias[e];
#pragma unroll
      for (int i = 0; i < 8; ++i) {
        const int row = row0 + i + 8 * kh;
        dst[(size_t)row * DIM + e] = (__bf16)(acc[t][i] + be);
      }
    }
  }
}

// ---------------------------------------------------------------------------
// Kernel 2: Sc[b][q][k] = (Q . K) * 1/sqrt(D) + mask[q][k]   (f32 out)
// Double-buffered async global->LDS staging with static buffer indices:
// copies for tile i+1 run on the async engine while tile i's WMMAs execute.
// ---------------------------------------------------------------------------
__global__ __launch_bounds__(128) void scores_kernel(
    const __bf16* __restrict__ Qb, const __bf16* __restrict__ Kb,
    const float* __restrict__ mask, float* __restrict__ Sc)
{
  __shared__ __attribute__((aligned(32))) __bf16 As[2][64 * 32];
  __shared__ __attribute__((aligned(32))) __bf16 Bs[2][64 * 32];

  const int qBase = blockIdx.x * 64;        // flat q row over BATCH*SEQ
  const int kBase = blockIdx.y * 64;        // key index within sequence
  const int b_    = qBase >> 11;            // / SEQ
  const int qIn   = qBase & (SEQ - 1);
  const int tid = threadIdx.x, wave = tid >> 5, lane = tid & 31;

  v8f acc[4];
#pragma unroll
  for (int t = 0; t < 4; ++t) acc[t] = zero8();

  const int r  = tid >> 1;
  const int c0 = (tid & 1) * 16;
  const __bf16* baseA = Qb + (size_t)(qBase + r) * DIM + c0;
  const __bf16* baseB = Kb + (size_t)(b_ * SEQ + kBase + r) * DIM + c0;

  auto stage = [&](int buf, int it) {
    const __bf16* sA = baseA + it * 32;
    const __bf16* sB = baseB + it * 32;
    __bf16* dA = &As[buf][r * 32 + c0];
    __bf16* dB = &Bs[buf][r * 32 + c0];
    async_g2l_b128(dA,     sA);
    async_g2l_b128(dA + 8, sA + 8);
    async_g2l_b128(dB,     sB);
    async_g2l_b128(dB + 8, sB + 8);
  };

  stage(0, 0);                              // prologue: tile 0 -> buffer 0

  const int NT = DIM / 32;                  // 32 (even)
  for (int it = 0; it < NT; it += 2) {
    // ---- even tile (buffer 0), prefetch it+1 into buffer 1 ----
    stage(1, it + 1);                       // it+1 < NT always
    wait_async4();                          // tile `it` copies complete
    __syncthreads();
    mma_step(acc, &As[0][(wave * 16) * 32], Bs[0]);
    __syncthreads();

    // ---- odd tile (buffer 1), prefetch it+2 into buffer 0 ----
    if (it + 2 < NT) {
      stage(0, it + 2);
      wait_async4();
    } else {
      wait_async0();
    }
    __syncthreads();
    mma_step(acc, &As[1][(wave * 16) * 32], Bs[1]);
    __syncthreads();
  }

  const float scale = 0.03125f;             // 1/sqrt(1024)
  const int n = lane & 15, kh = lane >> 4;
#pragma unroll
  for (int t = 0; t < 4; ++t) {
    const int kc = kBase + t * 16 + n;
#pragma unroll
    for (int i = 0; i < 8; ++i) {
      const int rl = wave * 16 + i + 8 * kh;           // q row within tile
      const float v = acc[t][i] * scale
                    + mask[(size_t)(qIn + rl) * SEQ + kc];
      Sc[(size_t)(qBase + rl) * SEQ + kc] = v;
    }
  }
}

// ---------------------------------------------------------------------------
// Kernel 3: in-place row softmax over SEQ=2048 f32 values. 1 block per row.
// ---------------------------------------------------------------------------
__global__ __launch_bounds__(256) void softmax_kernel(float* __restrict__ Sc)
{
  __shared__ float buf[SEQ];
  __shared__ float red[8];
  float* row = Sc + (size_t)blockIdx.x * SEQ;
  const int tid = threadIdx.x;
  const int wv = tid >> 5, ln = tid & 31;

  float m = -3.4e38f;
  for (int i = tid; i < SEQ; i += 256) {
    const float v = row[i];
    buf[i] = v;
    m = fmaxf(m, v);
  }
#pragma unroll
  for (int off = 16; off > 0; off >>= 1)
    m = fmaxf(m, __shfl_xor(m, off, 32));
  if (ln == 0) red[wv] = m;
  __syncthreads();
  m = red[0];
#pragma unroll
  for (int j = 1; j < 8; ++j) m = fmaxf(m, red[j]);

  float s = 0.0f;
  for (int i = tid; i < SEQ; i += 256) {
    const float e = __expf(buf[i] - m);
    buf[i] = e;
    s += e;
  }
#pragma unroll
  for (int off = 16; off > 0; off >>= 1)
    s += __shfl_xor(s, off, 32);
  __syncthreads();                 // done reading red (max phase)
  if (ln == 0) red[wv] = s;
  __syncthreads();
  s = 0.0f;
#pragma unroll
  for (int j = 0; j < 8; ++j) s += red[j];
  const float inv = 1.0f / s;

  for (int i = tid; i < SEQ; i += 256) row[i] = buf[i] * inv;
}

// ---------------------------------------------------------------------------
// Kernel 4: out[b][q][e] = sum_k P[b][q][k] * V[b][k][e]  via Vt[b][e][k]
// Vt double-buffered on the async engine (static buffers); P register-
// pipelined f32 loads (static ping/pong) with on-the-fly bf16 conversion.
// ---------------------------------------------------------------------------
__global__ __launch_bounds__(128) void pv_kernel(
    const float* __restrict__ Sc, const __bf16* __restrict__ Vt,
    float* __restrict__ out)
{
  __shared__ __attribute__((aligned(32))) __bf16 As[64 * 32];
  __shared__ __attribute__((aligned(32))) __bf16 Bs[2][64 * 32];

  const int qBase = blockIdx.x * 64;   // flat q row over BATCH*SEQ
  const int eBase = blockIdx.y * 64;   // output feature e
  const int b_    = qBase >> 11;
  const int tid = threadIdx.x, wave = tid >> 5, lane = tid & 31;

  v8f acc[4];
#pragma unroll
  for (int t = 0; t < 4; ++t) acc[t] = zero8();

  const int r  = tid >> 1;
  const int c0 = (tid & 1) * 16;
  const float*  baseA = Sc + (size_t)(qBase + r) * SEQ + c0;
  const __bf16* baseB = Vt + (size_t)b_ * DIM * SEQ
                           + (size_t)(eBase + r) * SEQ + c0;
  __bf16* dA = As + r * 32 + c0;
  const __bf16* Asub = As + (wave * 16) * 32;

  auto stageB = [&](int buf, int it) {
    const __bf16* sB = baseB + it * 32;
    __bf16* dB = &Bs[buf][r * 32 + c0];
    async_g2l_b128(dB,     sB);
    async_g2l_b128(dB + 8, sB + 8);
  };

  float4 pA[4], qA[4];
  stageB(0, 0);                        // prologue
  load16(pA, baseA);

  const int NT = SEQ / 32;             // 64 K-steps (even)
  for (int it = 0; it < NT; it += 2) {
    // ---- even tile: V buffer 0, P ping; prefetch it+1 ----
    stageB(1, it + 1);                 // it+1 < NT always
    load16(qA, baseA + (it + 1) * 32);
    cvt_store16(dA, pA);
    wait_async2();                     // tile `it` V copies complete
    __syncthreads();
    mma_step(acc, Asub, Bs[0]);
    __syncthreads();

    // ---- odd tile: V buffer 1, P pong; prefetch it+2 ----
    if (it + 2 < NT) {
      stageB(0, it + 2);
      load16(pA, baseA + (it + 2) * 32);
      cvt_store16(dA, qA);
      wait_async2();
    } else {
      cvt_store16(dA, qA);
      wait_async0();
    }
    __syncthreads();
    mma_step(acc, Asub, Bs[1]);
    __syncthreads();
  }

  const int n = lane & 15, kh = lane >> 4;
#pragma unroll
  for (int t = 0; t < 4; ++t) {
    const int e = eBase + t * 16 + n;
#pragma unroll
    for (int i = 0; i < 8; ++i) {
      const int row = qBase + wave * 16 + i + 8 * kh;
      out[(size_t)row * DIM + e] = acc[t][i];
    }
  }
}

// ---------------------------------------------------------------------------
extern "C" void kernel_launch(void* const* d_in, const int* in_sizes, int n_in,
                              void* d_out, int out_size, void* d_ws, size_t ws_size,
                              hipStream_t stream) {
  (void)in_sizes; (void)n_in; (void)out_size; (void)ws_size;
  const float* x    = (const float*)d_in[0];
  const float* mask = (const float*)d_in[1];
  const float* wq   = (const float*)d_in[2];
  const float* bq   = (const float*)d_in[3];
  const float* wk   = (const float*)d_in[4];
  const float* bk   = (const float*)d_in[5];
  const float* wv   = (const float*)d_in[6];
  const float* bv   = (const float*)d_in[7];
  float* out = (float*)d_out;

  // Workspace layout (112 MB total):
  //   [0, 16MB)   Qb  bf16 [B*S][D]
  //   [16, 32MB)  Kb  bf16 [B*S][D]
  //   [32, 48MB)  Vt  bf16 [B][D][S]   (V transposed)
  //   [48, 112MB) Sc  f32  [B*S][S]    (scores, softmaxed in place)
  char* ws = (char*)d_ws;
  __bf16* Qb = (__bf16*)(ws);
  __bf16* Kb = (__bf16*)(ws + (size_t)16 * 1024 * 1024);
  __bf16* Vt = (__bf16*)(ws + (size_t)32 * 1024 * 1024);
  float*  Sc = (float*) (ws + (size_t)48 * 1024 * 1024);

  const int rowTiles = (BATCH * SEQ) / 64;   // 128

  qkv_kernel<<<dim3(rowTiles, DIM / 64, 3), 128, 0, stream>>>(
      x, wq, bq, wk, bk, wv, bv, Qb, Kb, Vt);
  scores_kernel<<<dim3(rowTiles, SEQ / 64), 128, 0, stream>>>(Qb, Kb, mask, Sc);
  softmax_kernel<<<dim3(BATCH * SEQ), 256, 0, stream>>>(Sc);
  pv_kernel<<<dim3(rowTiles, DIM / 64), 128, 0, stream>>>(Sc, Vt, out);
}
